// GRUModel1_14826227106295
// MI455X (gfx1250) — compile-verified, matrix operand order checked by reading, not run on hardware
//
#include <hip/hip_runtime.h>
#include <hip/hip_bf16.h>

typedef __attribute__((ext_vector_type(16))) __bf16 v16bf;
typedef __attribute__((ext_vector_type(8)))  __bf16 v8bf;
typedef __attribute__((ext_vector_type(8)))  float  v8f;

namespace {
constexpr int HID     = 512;
constexpr int WELEMS  = 3 * HID * HID;    // 786432 elements per big weight matrix
constexpr int NSTEP   = 100;
constexpr int OUTD    = 3;
constexpr int OSTRIDE = NSTEP * OUTD;     // 300
constexpr int ROWS    = 32;               // batch rows per wave (two WMMA M-tiles)
constexpr int JTILES  = HID / 16;         // 32 output column tiles per gate
constexpr int KSTEPS  = HID / 32;         // 16 K-chunks per GEMM
constexpr int BATCH_N = 16384;
constexpr int CPY4    = ROWS * HID * 2 / 16;  // float4s per state buffer (2048)
}

union V16U { struct { v8bf lo; v8bf hi; } p; v16bf v; };

// Load one WMMA 16-bit operand (A or B) for this lane: two 16B chunks.
// Caller pre-applies the lane's kSel (0 or 8 elements); chunks sit at +0 and +16 elems.
static __device__ __forceinline__ v16bf ldmat(const __bf16* p) {
  V16U u;
  u.p.lo = *reinterpret_cast<const v8bf*>(p);
  u.p.hi = *reinterpret_cast<const v8bf*>(p + 16);
  return u.v;
}

static __device__ __forceinline__ v8f wmma_bf16(v16bf a, v16bf b, v8f c) {
  return __builtin_amdgcn_wmma_f32_16x16x32_bf16(false, a, false, b, (short)0, c,
                                                 false, false);
}

static __device__ __forceinline__ v8f vzero8() {
  v8f r = {0.f, 0.f, 0.f, 0.f, 0.f, 0.f, 0.f, 0.f};
  return r;
}

static __device__ __forceinline__ float sigf(float x) {
  return 1.0f / (1.0f + __expf(-x));
}
static __device__ __forceinline__ float tanh_f(float x) {
  float xc = fminf(fmaxf(x, -15.0f), 15.0f);
  float e  = __expf(2.0f * xc);
  return (e - 1.0f) / (e + 1.0f);
}

__global__ void cvt_bf16(const float* __restrict__ src, __bf16* __restrict__ dst, int n) {
  int i = blockIdx.x * blockDim.x + threadIdx.x;
  if (i < n) dst[i] = (__bf16)src[i];
}

__global__ __launch_bounds__(32, 1)
void gru_seq(const float* __restrict__ z,
             const float* __restrict__ w_ih1, const float* __restrict__ b_ih1,
             const float* __restrict__ b_hh1,
             const float* __restrict__ b_ih2, const float* __restrict__ b_hh2,
             const float* __restrict__ fc_w,  const float* __restrict__ fc_b,
             const __bf16* __restrict__ wh1,  const __bf16* __restrict__ wi2,
             const __bf16* __restrict__ wh2,  float* __restrict__ out)
{
  __shared__ __align__(16) __bf16 H1[ROWS * HID];   // h1 state (bf16)        32 KB
  __shared__ __align__(16) __bf16 H2[ROWS * HID];   // h2 state (bf16)        32 KB
  __shared__ __align__(16) __bf16 HT[ROWS * HID];   // double buffer h_new    32 KB
  __shared__ float X[ROWS * OUTD];                  // current x (f32)

  const int lane = threadIdx.x & 31;
  const int nCol = lane & 15;          // N col in C-tile; also row/col id for A/B loads
  const int mHi  = lane >> 4;          // 0 -> C rows v, 1 -> C rows v+8
  const int kSel = mHi * 8;            // lane-half K offset for 16-bit operand layout
  const int rowBase = blockIdx.x * ROWS;

  // zero-init hidden states
  {
    float4 z4 = make_float4(0.f, 0.f, 0.f, 0.f);
    float4* p1 = reinterpret_cast<float4*>(H1);
    float4* p2 = reinterpret_cast<float4*>(H2);
    for (int i = lane; i < CPY4; i += 32) { p1[i] = z4; p2[i] = z4; }
  }
  // x0 = z[:, 3:6]; emit t=0 output (one row per lane)
  {
    const int b = rowBase + lane;
    for (int d = 0; d < OUTD; ++d) {
      float xv = z[(size_t)b * 6 + 3 + d];
      X[lane * OUTD + d] = xv;
      out[(size_t)b * OSTRIDE + d] = xv;
    }
  }

  #pragma unroll 1
  for (int t = 1; t < NSTEP; ++t) {
    // x rows in C-tile layout for both M-halves (live only through layer 1)
    float xr[2][8][3];
    #pragma unroll
    for (int half = 0; half < 2; ++half)
      #pragma unroll
      for (int v = 0; v < 8; ++v)
        #pragma unroll
        for (int d = 0; d < OUTD; ++d)
          xr[half][v][d] = X[(half * 16 + v + 8 * mHi) * OUTD + d];

    // ---------------- layer 1: h1 = GRU(x, h1) ----------------
    #pragma unroll 1
    for (int jt = 0; jt < JTILES; ++jt) {
      v8f aR0 = vzero8(), aZ0 = vzero8(), aN0 = vzero8();
      v8f aR1 = vzero8(), aZ1 = vzero8(), aN1 = vzero8();
      const __bf16* bcol  = wh1 + (size_t)(jt * 16 + nCol) * HID + kSel;
      const __bf16* arow0 = H1 + nCol * HID + kSel;
      const __bf16* arow1 = H1 + (16 + nCol) * HID + kSel;
      #pragma unroll 4
      for (int kk = 0; kk < KSTEPS; ++kk) {
        v16bf a0 = ldmat(arow0 + kk * 32);
        v16bf a1 = ldmat(arow1 + kk * 32);
        v16bf bR = ldmat(bcol + kk * 32);
        v16bf bZ = ldmat(bcol + (size_t)HID * HID + kk * 32);
        v16bf bN = ldmat(bcol + (size_t)2 * HID * HID + kk * 32);
        aR0 = wmma_bf16(a0, bR, aR0);  aR1 = wmma_bf16(a1, bR, aR1);
        aZ0 = wmma_bf16(a0, bZ, aZ0);  aZ1 = wmma_bf16(a1, bZ, aZ1);
        aN0 = wmma_bf16(a0, bN, aN0);  aN1 = wmma_bf16(a1, bN, aN1);
      }
      const int j = jt * 16 + nCol;
      const float bir = b_ih1[j], biz = b_ih1[HID + j], bin_ = b_ih1[2 * HID + j];
      const float bhr = b_hh1[j], bhz = b_hh1[HID + j], bhn = b_hh1[2 * HID + j];
      float wir[3], wiz[3], win[3];
      #pragma unroll
      for (int d = 0; d < 3; ++d) {
        wir[d] = w_ih1[(size_t)j * 3 + d];
        wiz[d] = w_ih1[(size_t)(HID + j) * 3 + d];
        win[d] = w_ih1[(size_t)(2 * HID + j) * 3 + d];
      }
      #pragma unroll
      for (int half = 0; half < 2; ++half) {
        const v8f& gR = half ? aR1 : aR0;
        const v8f& gZ = half ? aZ1 : aZ0;
        const v8f& gN = half ? aN1 : aN0;
        #pragma unroll
        for (int v = 0; v < 8; ++v) {
          const int m = half * 16 + v + 8 * mHi;
          const float* xv = xr[half][v];
          float gir = bir + xv[0] * wir[0] + xv[1] * wir[1] + xv[2] * wir[2];
          float giz = biz + xv[0] * wiz[0] + xv[1] * wiz[1] + xv[2] * wiz[2];
          float gin = bin_ + xv[0] * win[0] + xv[1] * win[1] + xv[2] * win[2];
          float r  = sigf(gir + gR[v] + bhr);
          float zt = sigf(giz + gZ[v] + bhz);
          float nn = tanh_f(gin + r * (gN[v] + bhn));
          float ho = (float)H1[m * HID + j];
          HT[m * HID + j] = (__bf16)((1.f - zt) * nn + zt * ho);
        }
      }
    }
    { // commit: HT -> H1 (same-wave DS ordering guarantees correctness)
      const float4* s = reinterpret_cast<const float4*>(HT);
      float4* dp = reinterpret_cast<float4*>(H1);
      for (int i = lane; i < CPY4; i += 32) dp[i] = s[i];
    }

    // ---------------- layer 2: h2 = GRU(h1, h2) ----------------
    #pragma unroll 1
    for (int jt = 0; jt < JTILES; ++jt) {
      v8f iR0 = vzero8(), iZ0 = vzero8(), iN0 = vzero8();
      v8f hR0 = vzero8(), hZ0 = vzero8(), hN0 = vzero8();
      v8f iR1 = vzero8(), iZ1 = vzero8(), iN1 = vzero8();
      v8f hR1 = vzero8(), hZ1 = vzero8(), hN1 = vzero8();
      const __bf16* bi = wi2 + (size_t)(jt * 16 + nCol) * HID + kSel;
      const __bf16* bh = wh2 + (size_t)(jt * 16 + nCol) * HID + kSel;
      const __bf16* a1r0 = H1 + nCol * HID + kSel;
      const __bf16* a1r1 = H1 + (16 + nCol) * HID + kSel;
      const __bf16* a2r0 = H2 + nCol * HID + kSel;
      const __bf16* a2r1 = H2 + (16 + nCol) * HID + kSel;
      #pragma unroll 2
      for (int kk = 0; kk < KSTEPS; ++kk) {
        v16bf x10 = ldmat(a1r0 + kk * 32);
        v16bf x11 = ldmat(a1r1 + kk * 32);
        v16bf x20 = ldmat(a2r0 + kk * 32);
        v16bf x21 = ldmat(a2r1 + kk * 32);
        v16bf bR  = ldmat(bi + kk * 32);
        v16bf bZ  = ldmat(bi + (size_t)HID * HID + kk * 32);
        v16bf bN  = ldmat(bi + (size_t)2 * HID * HID + kk * 32);
        iR0 = wmma_bf16(x10, bR, iR0);  iR1 = wmma_bf16(x11, bR, iR1);
        iZ0 = wmma_bf16(x10, bZ, iZ0);  iZ1 = wmma_bf16(x11, bZ, iZ1);
        iN0 = wmma_bf16(x10, bN, iN0);  iN1 = wmma_bf16(x11, bN, iN1);
        v16bf cR  = ldmat(bh + kk * 32);
        v16bf cZ  = ldmat(bh + (size_t)HID * HID + kk * 32);
        v16bf cN  = ldmat(bh + (size_t)2 * HID * HID + kk * 32);
        hR0 = wmma_bf16(x20, cR, hR0);  hR1 = wmma_bf16(x21, cR, hR1);
        hZ0 = wmma_bf16(x20, cZ, hZ0);  hZ1 = wmma_bf16(x21, cZ, hZ1);
        hN0 = wmma_bf16(x20, cN, hN0);  hN1 = wmma_bf16(x21, cN, hN1);
      }
      const int j = jt * 16 + nCol;
      const float bir = b_ih2[j], biz = b_ih2[HID + j], bin_ = b_ih2[2 * HID + j];
      const float bhr = b_hh2[j], bhz = b_hh2[HID + j], bhn = b_hh2[2 * HID + j];
      #pragma unroll
      for (int half = 0; half < 2; ++half) {
        const v8f& giR = half ? iR1 : iR0;
        const v8f& giZ = half ? iZ1 : iZ0;
        const v8f& giN = half ? iN1 : iN0;
        const v8f& ghR = half ? hR1 : hR0;
        const v8f& ghZ = half ? hZ1 : hZ0;
        const v8f& ghN = half ? hN1 : hN0;
        #pragma unroll
        for (int v = 0; v < 8; ++v) {
          const int m = half * 16 + v + 8 * mHi;
          float r  = sigf(giR[v] + bir + ghR[v] + bhr);
          float zt = sigf(giZ[v] + biz + ghZ[v] + bhz);
          float nn = tanh_f(giN[v] + bin_ + r * (ghN[v] + bhn));
          float ho = (float)H2[m * HID + j];
          HT[m * HID + j] = (__bf16)((1.f - zt) * nn + zt * ho);
        }
      }
    }
    { // commit: HT -> H2
      const float4* s = reinterpret_cast<const float4*>(HT);
      float4* dp = reinterpret_cast<float4*>(H2);
      for (int i = lane; i < CPY4; i += 32) dp[i] = s[i];
    }

    // ---------------- head: x += 0.1 * tanh(h2 @ fc_w.T + fc_b) ----------------
    {
      float a0 = fc_b[0], a1v = fc_b[1], a2v = fc_b[2];
      const __bf16* hrow = H2 + lane * HID;
      #pragma unroll 4
      for (int k = 0; k < HID; k += 8) {
        v8bf hv = *reinterpret_cast<const v8bf*>(hrow + k);
        #pragma unroll
        for (int e = 0; e < 8; ++e) {
          float hf = (float)hv[e];
          a0  += hf * fc_w[k + e];
          a1v += hf * fc_w[HID + k + e];
          a2v += hf * fc_w[2 * HID + k + e];
        }
      }
      const int b = rowBase + lane;
      float x0 = X[lane * OUTD + 0] + 0.1f * tanh_f(a0);
      float x1 = X[lane * OUTD + 1] + 0.1f * tanh_f(a1v);
      float x2 = X[lane * OUTD + 2] + 0.1f * tanh_f(a2v);
      X[lane * OUTD + 0] = x0;
      X[lane * OUTD + 1] = x1;
      X[lane * OUTD + 2] = x2;
      float* op = out + (size_t)b * OSTRIDE + (size_t)t * OUTD;
      op[0] = x0; op[1] = x1; op[2] = x2;
    }
  }
}

extern "C" void kernel_launch(void* const* d_in, const int* in_sizes, int n_in,
                              void* d_out, int out_size, void* d_ws, size_t ws_size,
                              hipStream_t stream) {
  const float* z     = (const float*)d_in[0];
  const float* w_ih1 = (const float*)d_in[1];
  const float* w_hh1 = (const float*)d_in[2];
  const float* b_ih1 = (const float*)d_in[3];
  const float* b_hh1 = (const float*)d_in[4];
  const float* w_ih2 = (const float*)d_in[5];
  const float* w_hh2 = (const float*)d_in[6];
  const float* b_ih2 = (const float*)d_in[7];
  const float* b_hh2 = (const float*)d_in[8];
  const float* fc_w  = (const float*)d_in[9];
  const float* fc_b  = (const float*)d_in[10];

  __bf16* wh1 = (__bf16*)d_ws;            // 1536x512 bf16
  __bf16* wi2 = wh1 + WELEMS;
  __bf16* wh2 = wi2 + WELEMS;

  const int cb = (WELEMS + 255) / 256;
  cvt_bf16<<<cb, 256, 0, stream>>>(w_hh1, wh1, WELEMS);
  cvt_bf16<<<cb, 256, 0, stream>>>(w_ih2, wi2, WELEMS);
  cvt_bf16<<<cb, 256, 0, stream>>>(w_hh2, wh2, WELEMS);

  const int wgs = BATCH_N / ROWS;          // 512 single-wave workgroups x 32 rows
  gru_seq<<<wgs, 32, 0, stream>>>(z, w_ih1, b_ih1, b_hh1, b_ih2, b_hh2,
                                  fc_w, fc_b, wh1, wi2, wh2, (float*)d_out);
}
